// RelGraphConv_74302934221533
// MI455X (gfx1250) — matrix operand order repaired and checked by reading, not compiled, not run on hardware
//
#include <hip/hip_runtime.h>

typedef __attribute__((ext_vector_type(16))) _Float16 v16h;
typedef __attribute__((ext_vector_type(8)))  float    v8f;
typedef _Float16 f16;

#define DIN    128
#define DOUT   128
#define NBASE  8
#define KTOT   1152          // NBASE*DIN + DIN (self-loop fused into K)
#define KB     36            // KTOT / 32
#define NTILES 8             // DOUT / 16

// ---- WMMA f16 16x16x32 A-fragment swizzle (ISA 7.12.2, 16-bit A 16x32) ----
// lane L: row m = L&15, half h = L>>4 ; h=0 covers K in {0..7,16..23}, h=1 covers {8..15,24..31}
// element e of v16h: K = ((e>=8)?16:0) + (e&7 pattern)  => inverse used below.
__device__ __forceinline__ size_t a_swz(int mt, int k, int m) {
    int kb = k >> 5, r = k & 31;
    int h  = (r >> 3) & 1;                 // which half-lane group
    int e  = ((r & 16) >> 1) + (r & 7);    // element within v16h
    int lane = m + (h << 4);
    return ((((size_t)mt * KB) + kb) * 32 + lane) * 16 + e;
}

// ---------------- edge scatter: agg[N,R,128] += x[src]  (mode 0) ----------------
__global__ void scatter_nr_kernel(const float* __restrict__ x, const int* __restrict__ src,
                                  const int* __restrict__ dst, const int* __restrict__ et,
                                  float* __restrict__ agg, int E, int R) {
    int w = (blockIdx.x * blockDim.x + threadIdx.x) >> 5;   // one wave per edge
    if (w >= E) return;
    int lane = threadIdx.x & 31;
    int s = src[w], d = dst[w], r = et[w];
    const float4 v = ((const float4*)(x + (size_t)s * DIN))[lane];   // coalesced 512B row
    float* p = agg + ((size_t)d * R + r) * DIN + lane * 4;
    atomicAdd(p + 0, v.x); atomicAdd(p + 1, v.y);
    atomicAdd(p + 2, v.z); atomicAdd(p + 3, v.w);
}

// ---------------- edge scatter directly to bases (fallback, mode 1) ----------------
__global__ void scatter_nb_kernel(const float* __restrict__ x, const int* __restrict__ src,
                                  const int* __restrict__ dst, const int* __restrict__ et,
                                  const float* __restrict__ w_comp, float* __restrict__ aggB,
                                  int E) {
    int w = (blockIdx.x * blockDim.x + threadIdx.x) >> 5;
    if (w >= E) return;
    int lane = threadIdx.x & 31;
    int s = src[w], d = dst[w], r = et[w];
    const float4 v = ((const float4*)(x + (size_t)s * DIN))[lane];
#pragma unroll
    for (int b = 0; b < NBASE; ++b) {
        float c = w_comp[r * NBASE + b];
        float* p = aggB + ((size_t)d * NBASE + b) * DIN + lane * 4;
        atomicAdd(p + 0, c * v.x); atomicAdd(p + 1, c * v.y);
        atomicAdd(p + 2, c * v.z); atomicAdd(p + 3, c * v.w);
    }
}

// ---- fold relations->bases, convert to f16, write WMMA-swizzled A [N, KTOT] ----
__global__ void fold_kernel(const float* __restrict__ agg, const float* __restrict__ x,
                            const float* __restrict__ w_comp, f16* __restrict__ A,
                            int N, int R, int mode) {
    __shared__ float wc[256];
    int n = blockIdx.x;           // one node per block
    int i = threadIdx.x;          // channel 0..127
    for (int t = i; t < R * NBASE; t += blockDim.x) wc[t] = w_comp[t];
    __syncthreads();

    float tb[NBASE];
    if (mode == 0) {              // contract agg[n,R,128] over R with w_comp[R,8]
#pragma unroll
        for (int b = 0; b < NBASE; ++b) tb[b] = 0.f;
        const float* ar = agg + (size_t)n * R * DIN + i;
        for (int r = 0; r < R; ++r) {
            float a = ar[(size_t)r * DIN];
#pragma unroll
            for (int b = 0; b < NBASE; ++b) tb[b] = fmaf(a, wc[r * NBASE + b], tb[b]);
        }
    } else {                      // already folded per-edge
#pragma unroll
        for (int b = 0; b < NBASE; ++b) tb[b] = agg[((size_t)n * NBASE + b) * DIN + i];
    }

    int mt = n >> 4, m = n & 15;
#pragma unroll
    for (int b = 0; b < NBASE; ++b)
        A[a_swz(mt, b * DIN + i, m)] = (f16)tb[b];
    A[a_swz(mt, NBASE * DIN + i, m)] = (f16)x[(size_t)n * DIN + i];   // self-loop K-extension
}

// ---- pack weight(+loop_weight) f32 -> f16 in WMMA B-fragment layout ----
// B 32x16 tile: lanes 0-15 hold K={0..7,16..23} (col = lane&15), lanes 16-31 hold K={8..15,24..31}.
__global__ void wprep_kernel(const float* __restrict__ weight, const float* __restrict__ loopw,
                             f16* __restrict__ Bsw) {
    int t = blockIdx.x * blockDim.x + threadIdx.x;
    if (t >= NTILES * KB * 32 * 16) return;
    int e    = t & 15;
    int lane = (t >> 4) & 31;
    int rest = t >> 9;
    int kb   = rest % KB;
    int nb   = rest / KB;
    int o = nb * 16 + (lane & 15);
    int K = kb * 32 + ((e >> 3) << 4) + ((lane >> 4) << 3) + (e & 7);
    float v = (K < NBASE * DIN)
                  ? weight[(size_t)(K >> 7) * DIN * DOUT + (size_t)(K & 127) * DOUT + o]
                  : loopw[(size_t)(K - NBASE * DIN) * DOUT + o];
    Bsw[t] = (f16)v;
}

// ---- GEMM: out[N,128] = relu(A[N,1152] x W[1152,128] + bias), via v_wmma ----
__global__ void __launch_bounds__(256)
gemm_kernel(const f16* __restrict__ A, const f16* __restrict__ Bsw,
            const float* __restrict__ bias, float* __restrict__ out, int N) {
    int mt   = blockIdx.x;            // 16-node tile
    int nb   = threadIdx.x >> 5;      // 16-col tile (8 waves cover DOUT)
    int lane = threadIdx.x & 31;
    const v16h* Ap = (const v16h*)A   + ((size_t)mt * KB) * 32 + lane;
    const v16h* Bp = (const v16h*)Bsw + ((size_t)nb * KB) * 32 + lane;
    v8f acc = {};
#pragma unroll 4
    for (int kb = 0; kb < KB; ++kb) {
        v16h a = Ap[(size_t)kb * 32];
        v16h b = Bp[(size_t)kb * 32];
        acc = __builtin_amdgcn_wmma_f32_16x16x32_f16(false, a, false, b,
                                                     (short)0, acc, false, false);
    }
    int h   = lane >> 4;
    int col = nb * 16 + (lane & 15);
    float bo = bias[col];
#pragma unroll
    for (int j = 0; j < 8; ++j) {                 // C/D layout: row = j + h*8
        int node = mt * 16 + j + h * 8;
        if (node < N) out[(size_t)node * DOUT + col] = fmaxf(acc[j] + bo, 0.f);
    }
}

extern "C" void kernel_launch(void* const* d_in, const int* in_sizes, int n_in,
                              void* d_out, int out_size, void* d_ws, size_t ws_size,
                              hipStream_t stream) {
    const float* x      = (const float*)d_in[0];
    const int*   src    = (const int*)  d_in[1];
    const int*   dst    = (const int*)  d_in[2];
    const int*   et     = (const int*)  d_in[3];
    const float* weight = (const float*)d_in[4];
    const float* w_comp = (const float*)d_in[5];
    const float* h_bias = (const float*)d_in[6];
    const float* loopw  = (const float*)d_in[7];
    float*       out    = (float*)d_out;

    const int N = in_sizes[0] / DIN;
    const int E = in_sizes[1];
    const int R = in_sizes[5] / NBASE;            // 32
    const int mtiles = (N + 15) / 16;

    const size_t aggC_bytes = (size_t)N * R * DIN * sizeof(float);      // [N,R,128]
    const size_t aggB_bytes = (size_t)N * NBASE * DIN * sizeof(float);  // [N,8,128]
    const size_t A_bytes    = (size_t)mtiles * KB * 32 * 16 * sizeof(f16);
    const size_t B_bytes    = (size_t)NTILES * KB * 32 * 16 * sizeof(f16);

    int mode; size_t agg_bytes;
    if (ws_size >= aggC_bytes + A_bytes + B_bytes) { mode = 0; agg_bytes = aggC_bytes; }
    else                                           { mode = 1; agg_bytes = aggB_bytes; }

    char* wsp = (char*)d_ws;
    float* agg = (float*)wsp;
    f16*   A   = (f16*)(wsp + agg_bytes);
    f16*   Bsw = (f16*)(wsp + agg_bytes + A_bytes);

    hipMemsetAsync(agg, 0, agg_bytes, stream);

    int wtot = NTILES * KB * 32 * 16;
    wprep_kernel<<<(wtot + 255) / 256, 256, 0, stream>>>(weight, loopw, Bsw);

    unsigned gridE = (unsigned)(((size_t)E * 32 + 255) / 256);
    if (mode == 0)
        scatter_nr_kernel<<<gridE, 256, 0, stream>>>(x, src, dst, et, agg, E, R);
    else
        scatter_nb_kernel<<<gridE, 256, 0, stream>>>(x, src, dst, et, w_comp, agg, E);

    fold_kernel<<<N, DIN, 0, stream>>>(agg, x, w_comp, A, N, R, mode);

    gemm_kernel<<<mtiles, 256, 0, stream>>>(A, Bsw, h_bias, out, N);
}
